// RefMoE_154618823292
// MI455X (gfx1250) — matrix operand order repaired and checked by reading, not compiled, not run on hardware
//
#include <hip/hip_runtime.h>
#include <cstdint>
#include <cstddef>

typedef __attribute__((ext_vector_type(16))) __bf16 v16bf;
typedef __attribute__((ext_vector_type(8)))  float  v8f;

namespace {

constexpr int E_c = 8, H_c = 2048, I_c = 1408, SI_c = 2816, T_c = 4096;
constexpr int BM = 128, BN = 64, KB = 32;
constexpr int ARS = 20;   // A LDS row stride in dwords (16B-aligned, conflict-shifted)
constexpr int BCS = 20;   // B LDS column stride in dwords (fragment-major)

union FragB16 { uint4 q[2]; v16bf v; };

__device__ __forceinline__ unsigned short f2bf(float f) {
  unsigned u = __float_as_uint(f);
  u += 0x7FFFu + ((u >> 16) & 1u);          // round-to-nearest-even
  return (unsigned short)(u >> 16);
}
__device__ __forceinline__ unsigned pack2(float lo, float hi) {
  return (unsigned)f2bf(lo) | ((unsigned)f2bf(hi) << 16);
}

// CDNA5 async DMA: global -> LDS, 16B per lane, tracked by ASYNCcnt
__device__ __forceinline__ void async_g2l_b128(unsigned ldsOfs, const void* gaddr) {
  asm volatile("global_load_async_to_lds_b128 %0, %1, off"
               :: "v"(ldsOfs), "v"((unsigned long long)gaddr) : "memory");
}
__device__ __forceinline__ unsigned lds_ofs(const void* p) {
  return (unsigned)(unsigned long long)p;   // LDS aperture: addr[31:0] = LDS offset
}

// ---- f32 -> bf16 bulk convert (row-major kept): for x ------------------------
__global__ __launch_bounds__(256) void cvt_f32_to_bf16(const float* __restrict__ src,
                                                       unsigned* __restrict__ dst) {
  size_t i = (size_t)blockIdx.x * blockDim.x + threadIdx.x;
  const float4* s4 = (const float4*)src;
  float4 a = s4[2 * i], b = s4[2 * i + 1];
  uint4 o;
  o.x = pack2(a.x, a.y); o.y = pack2(a.z, a.w);
  o.z = pack2(b.x, b.y); o.w = pack2(b.z, b.w);
  ((uint4*)dst)[i] = o;
}

// ---- weight convert+pack: f32 [Kd][Nd] -> bf16 K-pair dwords [Nd][Kd/2] -----
__global__ __launch_bounds__(256) void cvt_pack_w(const float* __restrict__ src,
                                                  unsigned* __restrict__ dst,
                                                  int Kd, int Nd,
                                                  long long srcStrideE, long long dstStrideE) {
  __shared__ __align__(16) unsigned short lt[64][40];  // transposed tile [n][k], 80B rows
  const float* s = src + (size_t)blockIdx.z * (size_t)srcStrideE;
  unsigned*    d = dst + (size_t)blockIdx.z * (size_t)dstStrideE;
  const int k0 = blockIdx.y * 32, n0 = blockIdx.x * 64;
  const int tid = threadIdx.x;
  {
    int k = tid >> 3, n8 = (tid & 7) * 8;
    const float4* p = (const float4*)(s + (size_t)(k0 + k) * Nd + n0 + n8);
    float4 a = p[0], b = p[1];
    lt[n8 + 0][k] = f2bf(a.x); lt[n8 + 1][k] = f2bf(a.y);
    lt[n8 + 2][k] = f2bf(a.z); lt[n8 + 3][k] = f2bf(a.w);
    lt[n8 + 4][k] = f2bf(b.x); lt[n8 + 5][k] = f2bf(b.y);
    lt[n8 + 6][k] = f2bf(b.z); lt[n8 + 7][k] = f2bf(b.w);
  }
  __syncthreads();
  {
    int n = tid >> 2, seg = tid & 3;
    uint4 v = *(const uint4*)&lt[n][seg * 8];          // 8 ushorts = 4 K-pair dwords
    *(uint4*)(d + (size_t)(n0 + n) * (Kd / 2) + k0 / 2 + seg * 4) = v;
  }
}

// ---- routing: counting sort of (token,slot) pairs by expert -----------------
__global__ void route_zero(int* counts) { if ((int)threadIdx.x < E_c) counts[threadIdx.x] = 0; }

__global__ __launch_bounds__(256) void route_count(const int* __restrict__ idx,
                                                   int* __restrict__ counts) {
  int i = blockIdx.x * 256 + threadIdx.x;          // n = T*K = 8192 exact
  atomicAdd(&counts[idx[i]], 1);
}

__global__ void route_scan(const int* __restrict__ counts, int* __restrict__ offs,
                           int* __restrict__ cursor) {
  if (threadIdx.x == 0) {
    int s = 0;
    for (int e = 0; e < E_c; ++e) { offs[e] = s; cursor[e] = s; s += counts[e]; }
  }
}

__global__ __launch_bounds__(256) void route_scatter(const int* __restrict__ idx,
                                                     const float* __restrict__ tw,
                                                     int* __restrict__ cursor,
                                                     int* __restrict__ perm,
                                                     float* __restrict__ wgt,
                                                     int* __restrict__ inv) {
  int i = blockIdx.x * 256 + threadIdx.x;
  int e = idx[i];
  int p = atomicAdd(&cursor[e], 1);
  perm[p] = i;           // i = token*K + slot
  wgt[p]  = tw[i];
  inv[i]  = p;
}

// ---- tiled bf16 WMMA GEMM, double-buffered LDS, async-DMA staging -----------
// AMODE: 0 = plain rows, 1 = token-gather rows via perm (routed up),
//        2 = direct sorted rows with count clamp (routed down)
// SWIGLU: fuse silu(A@B[:, :half]) * (A@B[:, half:]), pairOff = column offset
// OMODE: 0 = bf16 store, 1 = f32 store
// NOTE: requires Kdim/KB even (true for all launches here: 64, 88, 44).
template <int AMODE, int SWIGLU, int OMODE>
__global__ __launch_bounds__(256) void gemm_bf16(
    const unsigned short* __restrict__ A, int lda,
    const unsigned* __restrict__ Bpk, long long bExpStride, int pairOff,
    void* __restrict__ Cout, int ldc, int Mtot, int Kdim,
    const int* __restrict__ counts, const int* __restrict__ offs,
    const int* __restrict__ perm) {
  __shared__ __align__(16) unsigned a_lds[2][BM * ARS];
  __shared__ __align__(16) unsigned b_lds[2][BN * BCS];
  __shared__ __align__(16) unsigned b2_lds[SWIGLU ? 2 : 1][SWIGLU ? BN * BCS : 4];

  const int tid = threadIdx.x;
  int base = 0, count = Mtot;
  if constexpr (AMODE != 0) {
    const int e = blockIdx.z;
    base = offs[e]; count = counts[e];
    if ((int)blockIdx.y * BM >= count) return;        // empty tile for this expert
  }
  const unsigned* __restrict__ Bp = Bpk + (size_t)blockIdx.z * (size_t)bExpStride;
  const int kd2 = Kdim >> 1;                          // packed column stride (dwords)
  const int n0 = blockIdx.x * BN;
  const int m0 = blockIdx.y * BM;

  const int wave = tid >> 5, lane = tid & 31;
  const int lm = lane & 15, lh = lane >> 4;

  int arow[2];
#pragma unroll
  for (int i = 0; i < 2; ++i) {
    int c = tid + i * 256;
    int r = c >> 2;
    if constexpr (AMODE == 0) {
      arow[i] = m0 + r;
    } else if constexpr (AMODE == 1) {
      int p = base + m0 + r;
      arow[i] = (p < base + count) ? (perm[p] >> 1) : (perm[base] >> 1);
    } else {
      int p = base + m0 + r;
      arow[i] = (p < base + count) ? p : base;
    }
  }
  const int ar = tid >> 2, as = tid & 3;              // A staging row/segment
  const int bn = tid >> 2, bsg = (tid & 3) * 4;       // B staging column/segment

  // per-thread LDS destinations for async DMA (both ping-pong buffers)
  unsigned lA0[2], lA1[2], lB[2], lB2[2];
#pragma unroll
  for (int b = 0; b < 2; ++b) {
    lA0[b] = lds_ofs(&a_lds[b][ar * ARS + as * 4]);
    lA1[b] = lds_ofs(&a_lds[b][(ar + 64) * ARS + as * 4]);
    lB[b]  = lds_ofs(&b_lds[b][bn * BCS + bsg]);
    if constexpr (SWIGLU) lB2[b] = lds_ofs(&b2_lds[b][bn * BCS + bsg]);
    else lB2[b] = 0;
  }

  v8f acc[4], acc2[4];
  const v8f vzero = {0.f, 0.f, 0.f, 0.f, 0.f, 0.f, 0.f, 0.f};
#pragma unroll
  for (int c = 0; c < 4; ++c) { acc[c] = vzero; acc2[c] = vzero; }

  // staging = pure async DMA global->LDS, no VGPR round-trip (ASYNCcnt)
  auto stageTile = [&](int buf, int kk) {
    async_g2l_b128(lA0[buf], A + (size_t)arow[0] * lda + kk + as * 8);
    async_g2l_b128(lA1[buf], A + (size_t)arow[1] * lda + kk + as * 8);
    const unsigned* bp = Bp + (size_t)(n0 + bn) * kd2 + (kk >> 1) + bsg;
    async_g2l_b128(lB[buf], bp);
    if constexpr (SWIGLU) async_g2l_b128(lB2[buf], bp + (size_t)pairOff * kd2);
  };
  // publish staged buffer: drain async DMA, then block barrier
  auto syncw = [&]() {
    asm volatile("s_wait_asynccnt 0" ::: "memory");
    __syncthreads();
  };

  const int am = wave * 16 + lm;

  auto computeTile = [&](int cur) {
    FragB16 af;
    {
      const unsigned* al = &a_lds[cur][am * ARS + lh * 8];
      af.q[0] = *(const uint4*)al;
      af.q[1] = *(const uint4*)(al + 4);
    }
    FragB16 bf[4];
#pragma unroll
    for (int c = 0; c < 4; ++c) {
      const unsigned* bl = &b_lds[cur][(c * 16 + lm) * BCS + lh * 8];
      bf[c].q[0] = *(const uint4*)bl;
      bf[c].q[1] = *(const uint4*)(bl + 4);
    }
    if constexpr (SWIGLU) {
      FragB16 b2[4];
#pragma unroll
      for (int c = 0; c < 4; ++c) {
        const unsigned* l = &b2_lds[cur][(c * 16 + lm) * BCS + lh * 8];
        b2[c].q[0] = *(const uint4*)l;
        b2[c].q[1] = *(const uint4*)(l + 4);
      }
#pragma unroll
      for (int c = 0; c < 4; ++c) {
        acc[c]  = __builtin_amdgcn_wmma_f32_16x16x32_bf16(false, af.v, false, bf[c].v,
                                                          (short)0, acc[c], false, false);
        acc2[c] = __builtin_amdgcn_wmma_f32_16x16x32_bf16(false, af.v, false, b2[c].v,
                                                          (short)0, acc2[c], false, false);
      }
    } else {
#pragma unroll
      for (int c = 0; c < 4; ++c)
        acc[c] = __builtin_amdgcn_wmma_f32_16x16x32_bf16(false, af.v, false, bf[c].v,
                                                         (short)0, acc[c], false, false);
    }
  };

  // ping-pong unrolled by 2: buffer indices are compile-time constants
  const int nIter = Kdim / KB;                   // even for all launches
  stageTile(0, 0);
  for (int it = 0; it < nIter - 2; it += 2) {
    syncw(); stageTile(1, (it + 1) * KB); computeTile(0);
    syncw(); stageTile(0, (it + 2) * KB); computeTile(1);
  }
  syncw(); stageTile(1, (nIter - 1) * KB); computeTile(0);
  syncw(); computeTile(1);

  // epilogue: C layout VGPR r -> (M = lh*8 + r, N = lm)
#pragma unroll
  for (int c = 0; c < 4; ++c) {
#pragma unroll
    for (int r = 0; r < 8; ++r) {
      const int ml = wave * 16 + lh * 8 + r;
      const int nn = n0 + c * 16 + lm;
      float vo = acc[c][r];
      if constexpr (SWIGLU) {
        float vb = acc2[c][r];
        vo = (vo / (1.0f + __expf(-vo))) * vb;   // silu(a)*b
      }
      size_t row; bool ok = true;
      if constexpr (AMODE == 0) {
        row = (size_t)(m0 + ml);
      } else {
        int p = base + m0 + ml;
        ok = (p < base + count);
        row = (size_t)p;
      }
      if (ok) {
        if constexpr (OMODE == 0)
          ((unsigned short*)Cout)[row * (size_t)ldc + nn] = f2bf(vo);
        else
          ((float*)Cout)[row * (size_t)ldc + nn] = vo;
      }
    }
  }
}

// ---- final combine: out[t] += w0*rout[p0] + w1*rout[p1] (deterministic) -----
__global__ __launch_bounds__(256) void combine_routed(float* __restrict__ out,
                                                      const float* __restrict__ rout,
                                                      const int* __restrict__ inv,
                                                      const float* __restrict__ wgt) {
  int t = blockIdx.y;
  int h4 = blockIdx.x * 256 + threadIdx.x;          // H/4 = 512, grid.x = 2
  int p0 = inv[2 * t], p1 = inv[2 * t + 1];
  float w0 = wgt[p0], w1 = wgt[p1];
  size_t oi = (size_t)t * (H_c / 4) + h4;
  float4 o  = ((float4*)out)[oi];
  float4 r0 = ((const float4*)rout)[(size_t)p0 * (H_c / 4) + h4];
  float4 r1 = ((const float4*)rout)[(size_t)p1 * (H_c / 4) + h4];
  o.x += w0 * r0.x + w1 * r1.x;
  o.y += w0 * r0.y + w1 * r1.y;
  o.z += w0 * r0.z + w1 * r1.z;
  o.w += w0 * r0.w + w1 * r1.w;
  ((float4*)out)[oi] = o;
}

} // namespace

extern "C" void kernel_launch(void* const* d_in, const int* in_sizes, int n_in,
                              void* d_out, int out_size, void* d_ws, size_t ws_size,
                              hipStream_t stream) {
  (void)in_sizes; (void)n_in; (void)out_size; (void)ws_size;
  const float* x   = (const float*)d_in[0];   // [T, H]
  const float* tw  = (const float*)d_in[1];   // [T, K]
  const float* W1  = (const float*)d_in[2];   // [E, H, 2I]
  const float* W2  = (const float*)d_in[3];   // [E, I, H]
  const float* Ws1 = (const float*)d_in[4];   // [H, 2SI]
  const float* Ws2 = (const float*)d_in[5];   // [SI, H]
  const int*   ti  = (const int*)d_in[6];     // [T, K]
  float* out = (float*)d_out;                 // [T, H]

  // workspace carve-up (256B aligned)
  char* w = (char*)d_ws;
  auto carve = [&](size_t bytes) { char* p = w; w += (bytes + 255) & ~(size_t)255; return p; };
  unsigned short* xb  = (unsigned short*)carve((size_t)T_c * H_c * 2);
  unsigned* w1p  = (unsigned*)carve((size_t)E_c * H_c * 2 * I_c * 2);   // [E][2I][H/2]
  unsigned* w2p  = (unsigned*)carve((size_t)E_c * I_c * H_c * 2);       // [E][H][I/2]
  unsigned* ws1p = (unsigned*)carve((size_t)H_c * 2 * SI_c * 2);        // [2SI][H/2]
  unsigned* ws2p = (unsigned*)carve((size_t)SI_c * H_c * 2);            // [H][SI/2]
  unsigned short* acts = (unsigned short*)carve((size_t)T_c * SI_c * 2);
  unsigned short* actr = (unsigned short*)carve((size_t)T_c * 2 * I_c * 2);
  float* rout   = (float*)carve((size_t)T_c * 2 * H_c * 4);
  int*   counts = (int*)carve(E_c * 4);
  int*   offs   = (int*)carve(E_c * 4);
  int*   cursor = (int*)carve(E_c * 4);
  int*   perm   = (int*)carve((size_t)T_c * 2 * 4);
  float* wgt    = (float*)carve((size_t)T_c * 2 * 4);
  int*   inv    = (int*)carve((size_t)T_c * 2 * 4);

  // 1) activation convert (row-major) + weight convert/pack ([N][K/2] dwords)
  cvt_f32_to_bf16<<<(T_c * H_c) / 2048, 256, 0, stream>>>(x, (unsigned*)xb);
  cvt_pack_w<<<dim3(2 * I_c / 64, H_c / 32, E_c), 256, 0, stream>>>(
      W1, w1p, H_c, 2 * I_c, (long long)H_c * 2 * I_c, (long long)I_c * H_c);
  cvt_pack_w<<<dim3(H_c / 64, I_c / 32, E_c), 256, 0, stream>>>(
      W2, w2p, I_c, H_c, (long long)I_c * H_c, (long long)H_c * (I_c / 2));
  cvt_pack_w<<<dim3(2 * SI_c / 64, H_c / 32, 1), 256, 0, stream>>>(
      Ws1, ws1p, H_c, 2 * SI_c, 0, 0);
  cvt_pack_w<<<dim3(H_c / 64, SI_c / 32, 1), 256, 0, stream>>>(
      Ws2, ws2p, SI_c, H_c, 0, 0);

  // 2) routing (counting sort by expert; replay-safe: counters rebuilt each call)
  route_zero<<<1, 32, 0, stream>>>(counts);
  route_count<<<(T_c * 2) / 256, 256, 0, stream>>>(ti, counts);
  route_scan<<<1, 32, 0, stream>>>(counts, offs, cursor);
  route_scatter<<<(T_c * 2) / 256, 256, 0, stream>>>(ti, tw, cursor, perm, wgt, inv);

  // 3) shared expert: up+swiglu then down (down writes every element of out)
  gemm_bf16<0, 1, 0><<<dim3(SI_c / BN, T_c / BM, 1), 256, 0, stream>>>(
      xb, H_c, ws1p, 0, SI_c, acts, SI_c, T_c, H_c, nullptr, nullptr, nullptr);
  gemm_bf16<0, 0, 1><<<dim3(H_c / BN, T_c / BM, 1), 256, 0, stream>>>(
      acts, SI_c, ws2p, 0, 0, out, H_c, T_c, SI_c, nullptr, nullptr, nullptr);

  // 4) routed experts on gathered rows (8192 total rows across experts)
  gemm_bf16<1, 1, 0><<<dim3(I_c / BN, (T_c * 2) / BM, E_c), 256, 0, stream>>>(
      xb, H_c, w1p, (long long)I_c * H_c, I_c, actr, I_c, 0, H_c,
      counts, offs, perm);
  gemm_bf16<2, 0, 1><<<dim3(H_c / BN, (T_c * 2) / BM, E_c), 256, 0, stream>>>(
      actr, I_c, w2p, (long long)H_c * (I_c / 2), 0, rout, H_c, 0, I_c,
      counts, offs, perm);

  // 5) deterministic weighted combine into out
  combine_routed<<<dim3(H_c / 1024, T_c, 1), 256, 0, stream>>>(out, rout, inv, wgt);
}
